// Upsampling_16930761081227
// MI455X (gfx1250) — matrix operand order repaired and checked by reading, not compiled, not run on hardware
//
#include <hip/hip_runtime.h>
#include <hip/hip_bf16.h>

typedef __attribute__((ext_vector_type(16))) _Float16 v16h;
typedef __attribute__((ext_vector_type(8)))  float    v8f;

#define B_    8
#define S_    256
#define H_    256
#define MEL_  1000
#define ROWS_ (B_ * S_)          // 2048

// ---------------------------------------------------------------------------
// CDNA5 wave32 WMMA fragment mappings (CDNA5 ISA §7.12.2):
//  16-bit A 16x32:  lane = m + ((k>>3)&1)*16 ; element j = ((k>>4)&1)*8 + (k&7)
//  16-bit B 32x16:  lane = ((k>>3)&1)*16 + n ; element j = ((k>>4)&1)*8 + (k&7)
//  f32  C/D 16x16:  col = lane&15, row = r + (lane>=16 ? 8 : 0)
// We stage tiles in *fragment order* (each lane's 16 halves contiguous, 32 B)
// so a fragment load is one aligned v16h LDS read (ds_load_b128 x2) instead of
// 16 scalar ds_load_u16.
// ---------------------------------------------------------------------------

// ---------------------------------------------------------------------------
// Generic WMMA GEMM:  C[M,N] (=|+=) A[M,K] @ W[K,N] + bias
// 8 waves/block, 64x32 output tile, K stepped by 32, f32->f16 staged in LDS.
// seg (power of two) / rshift: A row read = m+rshift, valid iff
// (m & (seg-1))+rshift in [0,seg)  -> K=3 'SAME' 1-D convs become 3
// accumulated GEMM passes with per-batch zero padding.  blockIdx.z + strides
// give batched GEMM.
// ---------------------------------------------------------------------------
__global__ __launch_bounds__(256)
void gemm_wmma(const float* __restrict__ A, int lda, long long strideA,
               const float* __restrict__ W, int ldw, long long strideW,
               const float* __restrict__ bias,
               float* __restrict__ C, int ldc, long long strideC,
               int M, int N, int K, int seg, int rshift, int accum) {
  A += strideA * blockIdx.z;
  W += strideW * blockIdx.z;
  C += strideC * blockIdx.z;
  const int m0 = blockIdx.x * 64;
  const int n0 = blockIdx.y * 32;
  __shared__ alignas(32) _Float16 Af[4][32][16];  // 4 M-subtiles, frag order
  __shared__ alignas(32) _Float16 Bf[2][32][16];  // 2 N-subtiles, frag order
  const int tid   = threadIdx.x;
  const int wave  = tid >> 5;
  const int lane  = tid & 31;
  const int tm    = wave & 3;        // M-subtile index (16 rows each)
  const int tn    = wave >> 2;       // N-subtile index (16 cols each)
  const int smask = seg - 1;         // seg is a power of two
  v8f acc = {};
  for (int kt = 0; kt < K; kt += 32) {
    for (int e = tid; e < 64 * 32; e += 256) {
      const int r = e >> 5, k = e & 31;
      const int m = m0 + r;
      float v = 0.f;
      if (m < M && (kt + k) < K) {
        const int sm = (m & smask) + rshift;
        if (sm >= 0 && sm < seg) v = A[(long long)(m + rshift) * lda + kt + k];
      }
      Af[r >> 4][(r & 15) + (((k >> 3) & 1) << 4)]
        [(((k >> 4) & 1) << 3) + (k & 7)] = (_Float16)v;
    }
    for (int e = tid; e < 32 * 32; e += 256) {
      const int k = e >> 5, nn = e & 31;
      const int n = n0 + nn;
      const float v =
          ((kt + k) < K && n < N) ? W[(long long)(kt + k) * ldw + n] : 0.f;
      Bf[nn >> 4][(((k >> 3) & 1) << 4) + (nn & 15)][(((k >> 4) & 1) << 3) + (k & 7)] = (_Float16)v;
    }
    // speculative prefetch of next A tile rows -> global_prefetch_b8
    if (kt + 32 < K && tid < 64) {
      const int m = m0 + tid;
      if (m < M) __builtin_prefetch(&A[(long long)m * lda + kt + 32], 0, 1);
    }
    __syncthreads();
    const v16h a = *(const v16h*)(&Af[tm][lane][0]);
    const v16h b = *(const v16h*)(&Bf[tn][lane][0]);
    acc = __builtin_amdgcn_wmma_f32_16x16x32_f16(false, a, false, b, (short)0,
                                                 acc, false, false);
    __syncthreads();
  }
  const int col   = n0 + tn * 16 + (lane & 15);
  const int rbase = (lane >> 4) * 8;
  if (col < N) {
    const float bv = bias ? bias[col] : 0.f;
#pragma unroll
    for (int r = 0; r < 8; ++r) {
      const int m = m0 + tm * 16 + rbase + r;
      if (m < M) {
        const float o = acc[r] + bv;
        if (accum) C[(long long)m * ldc + col] += o;
        else       C[(long long)m * ldc + col]  = o;
      }
    }
  }
}

// ------------------------- small elementwise kernels -----------------------
__global__ void k_concat(const float* __restrict__ ce,
                         const float* __restrict__ spk,
                         float* __restrict__ xin) {
  const int i = blockIdx.x * 256 + threadIdx.x;
  if (i >= ROWS_ * 256) return;
  const int row = i >> 8, c = i & 255;
  const int b = row >> 8;
  xin[i] = (c < 192) ? ce[row * 192 + c] : spk[b * 64 + (c - 192)];
}

__global__ __launch_bounds__(256)
void k_ln256(const float* __restrict__ X, float* __restrict__ Y,
             const float* __restrict__ g, const float* __restrict__ bv) {
  __shared__ float red[256];
  const int row = blockIdx.x, t = threadIdx.x;
  const float x = X[row * 256 + t];
  red[t] = x; __syncthreads();
  for (int o = 128; o; o >>= 1) { if (t < o) red[t] += red[t + o]; __syncthreads(); }
  const float mean = red[0] * (1.f / 256.f); __syncthreads();
  const float d = x - mean;
  red[t] = d * d; __syncthreads();
  for (int o = 128; o; o >>= 1) { if (t < o) red[t] += red[t + o]; __syncthreads(); }
  const float var = red[0] * (1.f / 256.f);
  Y[row * 256 + t] = d * rsqrtf(var + 1e-5f) * g[t] + bv[t];
}

__global__ void k_glu(const float* __restrict__ h,
                      const unsigned char* __restrict__ mask,
                      float* __restrict__ out) {
  const int i = blockIdx.x * 256 + threadIdx.x;
  if (i >= ROWS_ * 256) return;
  const int row = i >> 8, c = i & 255;
  const float a = h[row * 512 + c];
  const float z = h[row * 512 + 256 + c];
  out[i] = mask[row] ? 0.f : a / (1.f + __expf(-z));
}

__global__ void k_dwconv(const float* __restrict__ g,
                         const float* __restrict__ dw,
                         float* __restrict__ out) {
  const int i = blockIdx.x * 256 + threadIdx.x;
  if (i >= ROWS_ * 256) return;
  const int row = i >> 8, c = i & 255, s = row & 255;
  const float w0 = dw[c * 3 + 0], w1 = dw[c * 3 + 1], w2 = dw[c * 3 + 2];
  const float mx = fmaxf(w0, fmaxf(w1, w2));
  const float e0 = __expf(w0 - mx), e1 = __expf(w1 - mx), e2 = __expf(w2 - mx);
  const float inv = 1.f / (e0 + e1 + e2);
  float acc = e1 * inv * g[i];
  if (s > 0)   acc += e0 * inv * g[i - 256];
  if (s < 255) acc += e2 * inv * g[i + 256];
  out[i] = acc;
}

__global__ void k_copy(const float* __restrict__ a, float* __restrict__ b, int n) {
  const int i = blockIdx.x * 256 + threadIdx.x;
  if (i < n) b[i] = a[i];
}

__global__ void k_maskrows(float* __restrict__ x,
                           const unsigned char* __restrict__ mask) {
  const int i = blockIdx.x * 256 + threadIdx.x;
  if (i >= ROWS_ * 256) return;
  if (mask[i >> 8]) x[i] = 0.f;
}

__global__ __launch_bounds__(256)
void k_p1(const float* __restrict__ V, const float* __restrict__ w,
          const float* __restrict__ b, const unsigned char* __restrict__ mask,
          float* __restrict__ D) {
  __shared__ float red[256];
  const int row = blockIdx.x, t = threadIdx.x;
  red[t] = V[row * 256 + t] * w[t]; __syncthreads();
  for (int o = 128; o; o >>= 1) { if (t < o) red[t] += red[t + o]; __syncthreads(); }
  if (t == 0) {
    const float z = red[0] + b[0];
    const float sp = (z > 20.f) ? z : log1pf(__expf(z));
    D[row] = mask[row] ? 0.f : sp;
  }
}

// per-batch: alpha, sD scan -> e_k/s_k, mel_len, mel_mask, sV/sT scaling
__global__ __launch_bounds__(256)
void k_scalars(const float* __restrict__ D, const float* __restrict__ sched,
               const float* __restrict__ rhythm,
               const float* __restrict__ V, const float* __restrict__ T,
               float* __restrict__ ek, float* __restrict__ sk,
               float* __restrict__ sV, float* __restrict__ sT,
               int* __restrict__ mel_i, float* __restrict__ out_alpha,
               float* __restrict__ out_mellen, float* __restrict__ out_melmask) {
  const int b = blockIdx.x, t = threadIdx.x;
  __shared__ float red[256];
  __shared__ float sc[256];
  const float d = D[b * 256 + t];
  red[t] = d; __syncthreads();
  for (int o = 128; o; o >>= 1) { if (t < o) red[t] += red[t + o]; __syncthreads(); }
  const float sumD = red[0]; __syncthreads();
  red[t] = sched[b * 256 + t]; __syncthreads();
  for (int o = 128; o; o >>= 1) { if (t < o) red[t] += red[t + o]; __syncthreads(); }
  const float sumS = red[0]; __syncthreads();
  const float alpha = (sumS / rhythm[b]) / sumD;
  const float sd = alpha * d;
  sc[t] = sd; __syncthreads();
  for (int o = 1; o < 256; o <<= 1) {          // inclusive Hillis-Steele scan
    const float prev = (t >= o) ? sc[t - o] : 0.f;
    __syncthreads();
    sc[t] += prev;
    __syncthreads();
  }
  const float e = sc[t];
  ek[b * 256 + t] = e;
  sk[b * 256 + t] = e - sd;
  int mel = (int)rintf(alpha * sumD);
  mel = mel < 1 ? 1 : (mel > MEL_ ? MEL_ : mel);
  if (t == 0) { mel_i[b] = mel; out_alpha[b] = alpha; out_mellen[b] = (float)mel; }
  for (int i = t; i < MEL_; i += 256)
    out_melmask[b * MEL_ + i] = (i >= mel) ? 1.f : 0.f;
  const long long base = (long long)b * 65536;
  for (int i = t; i < 65536; i += 256) {
    sV[base + i] = alpha * V[base + i];
    sT[base + i] = alpha * T[base + i];
  }
}

__global__ void k_reluln16(const float* __restrict__ X, float* __restrict__ Y,
                           const float* __restrict__ g,
                           const float* __restrict__ b) {
  const int row = blockIdx.x * blockDim.x + threadIdx.x;
  if (row >= ROWS_) return;
  float v[16]; float s = 0.f;
#pragma unroll
  for (int c = 0; c < 16; ++c) {
    float x = X[row * 16 + c];
    x = x > 0.f ? x : 0.f;
    v[c] = x; s += x;
  }
  const float m = s * (1.f / 16.f);
  float var = 0.f;
#pragma unroll
  for (int c = 0; c < 16; ++c) { const float d = v[c] - m; var += d * d; }
  const float inv = rsqrtf(var * (1.f / 16.f) + 1e-5f);
#pragma unroll
  for (int c = 0; c < 16; ++c) Y[row * 16 + c] = (v[c] - m) * inv * g[c] + b[c];
}

// ---------------------------------------------------------------------------
// conv1 (implicit GEMM, WMMA): feat generated on the fly.
// feat c0=Sm, c1=Em (zeroed only where mel_mask AND text_mask), c>=2 = Cv.
// M = 16 s-positions, N = 16 out channels, K = 3*3*18=162 padded to 192.
// grid (S/16=16, 1000/8=125, B); wave w handles mel row blockIdx.y*8+w.
// Weight fragments staged once per block in fragment order (v16h loads).
// ---------------------------------------------------------------------------
__global__ __launch_bounds__(256)
void k_conv1_wmma(const float* __restrict__ sk, const float* __restrict__ ek,
                  const float* __restrict__ Cv,
                  const unsigned char* __restrict__ tmask,
                  const int* __restrict__ mel_i,
                  const float* __restrict__ wk, const float* __restrict__ wb,
                  _Float16* __restrict__ h1) {
  __shared__ alignas(32) _Float16 Bwf[6][32][16];   // frag order per K-step
  __shared__ float    bsh[16];
  __shared__ _Float16 patch[8][3 * 18 * 18];
  const int tid  = threadIdx.x;
  const int s0   = blockIdx.x * 16;
  const int b    = blockIdx.z;
  const int wave = tid >> 5;
  const int lane = tid & 31;
  const int mel  = blockIdx.y * 8 + wave;
  const int mlen = mel_i[b];
  for (int e = tid; e < 6 * 32 * 16; e += 256) {
    const int kg = e >> 4;                 // global k = c + 18*kw + 54*kh
    const int o  = e & 15;
    const int kl = kg & 31;
    Bwf[kg >> 5][(((kl >> 3) & 1) << 4) + o][(((kl >> 4) & 1) << 3) + (kl & 7)] =
        (_Float16)((kg < 162) ? wk[kg * 16 + o] : 0.f);
  }
  if (tid < 16) bsh[tid] = wb[tid];
  _Float16* P = &patch[wave][0];          // [kh][si][c], si 0..17 <-> s0-1..s0+16
  for (int e = lane; e < 972; e += 32) {
    const int kh  = e / 324;
    const int rem = e - kh * 324;
    const int si  = rem / 18;
    const int c   = rem - si * 18;
    const int sg  = s0 + si - 1;
    const int mr  = mel + kh - 1;
    float v = 0.f;
    if (sg >= 0 && sg < 256 && mr >= 0 && mr < MEL_) {
      if (c >= 2) {
        v = Cv[(b * 256 + sg) * 16 + (c - 2)];            // mel-invariant
      } else {
        const bool am = (mr >= mlen) && tmask[b * 256 + sg];  // attn_mask
        if (!am) {
          const float t = (float)(mr + 1);
          v = (c == 0) ? (t - sk[b * 256 + sg]) : (ek[b * 256 + sg] - t);
        }
      }
    }
    P[e] = (_Float16)v;
  }
  __syncthreads();
  v8f acc = {};
  const int m     = lane & 15;
  const int kbase = (lane >> 4) << 3;
#pragma unroll
  for (int st = 0; st < 6; ++st) {
    v16h a;
#pragma unroll
    for (int v = 0; v < 8; ++v) {
      const int k0 = 32 * st + kbase + ((v < 4) ? (2 * v) : (16 + 2 * (v - 4)));
#pragma unroll
      for (int h = 0; h < 2; ++h) {
        const int k = k0 + h;
        _Float16 val = (_Float16)0.f;
        if (k < 162) {
          const int kh = k / 54;
          const int r  = k - kh * 54;
          const int kw = r / 18;
          const int c  = r - kw * 18;
          val = P[(kh * 18 + (m + kw)) * 18 + c];
        }
        a[2 * v + h] = val;
      }
    }
    const v16h bf = *(const v16h*)(&Bwf[st][lane][0]);
    acc = __builtin_amdgcn_wmma_f32_16x16x32_f16(false, a, false, bf, (short)0,
                                                 acc, false, false);
  }
  const int o     = lane & 15;
  const int rbase = (lane >> 4) * 8;
  const float bb  = bsh[o];
#pragma unroll
  for (int r = 0; r < 8; ++r) {
    const int s = s0 + rbase + r;
    const float x = acc[r] + bb;
    const float y = x / (1.f + __expf(-x));               // silu
    h1[(((long long)b * MEL_ + mel) * 256 + s) * 16 + o] = (_Float16)y;
  }
}

// conv2 + lw projection fused: K = 3*3*16=144 padded to 160; after the WMMA,
// silu + dot with lw_w, cross-lane reduce (16-lane halves) -> Wsc row scores.
__global__ __launch_bounds__(256)
void k_conv2_wmma(const _Float16* __restrict__ h1,
                  const float* __restrict__ wk, const float* __restrict__ wb,
                  const float* __restrict__ lww, const float* __restrict__ lwb,
                  float* __restrict__ Wsc) {
  __shared__ alignas(32) _Float16 Bwf[5][32][16];
  __shared__ float    bsh[16];
  __shared__ float    lsh[16];
  __shared__ float    lb;
  __shared__ _Float16 patch[8][3 * 18 * 16];
  const int tid  = threadIdx.x;
  const int s0   = blockIdx.x * 16;
  const int b    = blockIdx.z;
  const int wave = tid >> 5;
  const int lane = tid & 31;
  const int mel  = blockIdx.y * 8 + wave;
  for (int e = tid; e < 5 * 32 * 16; e += 256) {
    const int kg = e >> 4;                 // global k = c + 16*kw + 48*kh
    const int o  = e & 15;
    const int kl = kg & 31;
    Bwf[kg >> 5][(((kl >> 3) & 1) << 4) + o][(((kl >> 4) & 1) << 3) + (kl & 7)] =
        (_Float16)((kg < 144) ? wk[kg * 16 + o] : 0.f);
  }
  if (tid < 16) { bsh[tid] = wb[tid]; lsh[tid] = lww[tid]; }
  if (tid == 0) lb = lwb[0];
  _Float16* P = &patch[wave][0];
  for (int e = lane; e < 864; e += 32) {
    const int kh  = e / 288;
    const int rem = e - kh * 288;
    const int si  = rem >> 4;
    const int c   = rem & 15;
    const int sg  = s0 + si - 1;
    const int mr  = mel + kh - 1;
    _Float16 v = (_Float16)0.f;
    if (sg >= 0 && sg < 256 && mr >= 0 && mr < MEL_)
      v = h1[(((long long)b * MEL_ + mr) * 256 + sg) * 16 + c];
    P[e] = v;
  }
  __syncthreads();
  v8f acc = {};
  const int m     = lane & 15;
  const int kbase = (lane >> 4) << 3;
#pragma unroll
  for (int st = 0; st < 5; ++st) {
    v16h a;
#pragma unroll
    for (int v = 0; v < 8; ++v) {
      const int k0 = 32 * st + kbase + ((v < 4) ? (2 * v) : (16 + 2 * (v - 4)));
#pragma unroll
      for (int h = 0; h < 2; ++h) {
        const int k = k0 + h;
        _Float16 val = (_Float16)0.f;
        if (k < 144) {
          const int kh = k / 48;
          const int r  = k - kh * 48;
          const int kw = r >> 4;
          const int c  = r & 15;
          val = P[(kh * 18 + (m + kw)) * 16 + c];
        }
        a[2 * v + h] = val;
      }
    }
    const v16h bf = *(const v16h*)(&Bwf[st][lane][0]);
    acc = __builtin_amdgcn_wmma_f32_16x16x32_f16(false, a, false, bf, (short)0,
                                                 acc, false, false);
  }
  const int o     = lane & 15;
  const int rbase = (lane >> 4) * 8;
  float vals[8];
#pragma unroll
  for (int r = 0; r < 8; ++r) {
    const float x = acc[r] + bsh[o];
    const float y = x / (1.f + __expf(-x));
    vals[r] = y * lsh[o];
  }
#pragma unroll
  for (int off = 1; off < 16; off <<= 1)
#pragma unroll
    for (int r = 0; r < 8; ++r) vals[r] += __shfl_xor(vals[r], off, 32);
  if (o == 0) {
#pragma unroll
    for (int r = 0; r < 8; ++r) {
      const int s = s0 + rbase + r;
      Wsc[((long long)b * MEL_ + mel) * 256 + s] = vals[r] + lb;
    }
  }
}

// softmax over S (mask collapses to text_mask since attn|text == text)
__global__ __launch_bounds__(256)
void k_softmax(const float* __restrict__ Wsc,
               const unsigned char* __restrict__ tmask,
               float* __restrict__ Wout, float* __restrict__ Wws) {
  __shared__ float red[256];
  const int mel = blockIdx.x, b = blockIdx.y, t = threadIdx.x;
  const long long base = ((long long)b * MEL_ + mel) * 256;
  const bool m = tmask[b * 256 + t] != 0;
  const float x = m ? -3.4e38f : Wsc[base + t];
  red[t] = x; __syncthreads();
  for (int o = 128; o; o >>= 1) { if (t < o) red[t] = fmaxf(red[t], red[t + o]); __syncthreads(); }
  const float mx = red[0]; __syncthreads();
  const float e = m ? 0.f : __expf(x - mx);
  red[t] = e; __syncthreads();
  for (int o = 128; o; o >>= 1) { if (t < o) red[t] += red[t + o]; __syncthreads(); }
  const float w = e / red[0];
  Wout[base + t] = w;
  Wws[base + t]  = w;
}

__global__ __launch_bounds__(256)
void k_upln(const float* __restrict__ up, const float* __restrict__ g,
            const float* __restrict__ bv, const int* __restrict__ mel_i,
            float* __restrict__ out) {
  __shared__ float red[256];
  const int mel = blockIdx.x, b = blockIdx.y, t = threadIdx.x;
  const long long base = ((long long)b * MEL_ + mel) * 256;
  const float x = up[base + t];
  red[t] = x; __syncthreads();
  for (int o = 128; o; o >>= 1) { if (t < o) red[t] += red[t + o]; __syncthreads(); }
  const float mean = red[0] * (1.f / 256.f); __syncthreads();
  const float d = x - mean;
  red[t] = d * d; __syncthreads();
  for (int o = 128; o; o >>= 1) { if (t < o) red[t] += red[t + o]; __syncthreads(); }
  const float var = red[0] * (1.f / 256.f);
  const float y = d * rsqrtf(var + 1e-5f) * g[t] + bv[t];
  out[base + t] = (mel >= mel_i[b]) ? 0.f : y;
}

// ---------------------------------------------------------------------------
extern "C" void kernel_launch(void* const* d_in, const int* in_sizes, int n_in,
                              void* d_out, int out_size, void* d_ws,
                              size_t ws_size, hipStream_t stream) {
  (void)in_sizes; (void)n_in; (void)out_size; (void)ws_size;
  const float* ce     = (const float*)d_in[0];
  const float* spk    = (const float*)d_in[1];
  const float* rhythm = (const float*)d_in[2];
  const float* sched  = (const float*)d_in[3];
  const unsigned char* tmask = (const unsigned char*)d_in[4];
  const float* in_w = (const float*)d_in[5];
  const float* in_b = (const float*)d_in[6];
  const float* p1_w = (const float*)d_in[21];
  const float* p1_b = (const float*)d_in[22];
  const float* p2_k = (const float*)d_in[23];
  const float* p2_b = (const float*)d_in[24];
  const float* cw_k = (const float*)d_in[25];
  const float* cw_b = (const float*)d_in[26];
  const float* cw_ln_g = (const float*)d_in[27];
  const float* cw_ln_b = (const float*)d_in[28];
  const float* mkb1_k = (const float*)d_in[29];
  const float* mkb1_b = (const float*)d_in[30];
  const float* mkb2_k = (const float*)d_in[31];
  const float* mkb2_b = (const float*)d_in[32];
  const float* lw_w = (const float*)d_in[33];
  const float* lw_b = (const float*)d_in[34];
  const float* ln_g = (const float*)d_in[35];
  const float* ln_b = (const float*)d_in[36];

  // workspace carve-out
  char* wsb = (char*)d_ws;
  size_t off = 0;
  auto alloc = [&](size_t bytes) -> void* {
    void* p = wsb + off;
    off = (off + bytes + 255) & ~(size_t)255;
    return p;
  };
  const size_t RC = (size_t)ROWS_ * 256;            // 524288
  float* xin   = (float*)alloc(RC * 4);
  float* x0    = (float*)alloc(RC * 4);
  float* V1    = (float*)alloc(RC * 4);
  float* V2    = (float*)alloc(RC * 4);
  float* lnbuf = (float*)alloc(RC * 4);
  float* gbuf  = (float*)alloc(RC * 4);
  float* g2buf = (float*)alloc(RC * 4);
  float* T     = (float*)alloc(RC * 4);
  float* sV    = (float*)alloc(RC * 4);
  float* sT    = (float*)alloc(RC * 4);
  float* hglu  = (float*)alloc(RC * 2 * 4);
  float* Dv    = (float*)alloc(ROWS_ * 4);
  float* ek    = (float*)alloc(ROWS_ * 4);
  float* sk    = (float*)alloc(ROWS_ * 4);
  float* Cvraw = (float*)alloc(ROWS_ * 16 * 4);
  float* Cv    = (float*)alloc(ROWS_ * 16 * 4);
  int*   mel_i = (int*)alloc(64);
  _Float16* h1 = (_Float16*)alloc((size_t)B_ * MEL_ * 256 * 16 * 2);
  float* Wsc   = (float*)alloc((size_t)B_ * MEL_ * 256 * 4);
  float* upraw = (float*)alloc((size_t)B_ * MEL_ * 256 * 4);

  float* out      = (float*)d_out;
  float* out_up   = out;
  float* out_W    = out + (size_t)B_ * MEL_ * 256;
  float* out_a    = out + (size_t)2 * B_ * MEL_ * 256;
  float* out_ml   = out_a + B_;
  float* out_mm   = out_ml + B_;

  const int BIG = 1 << 30;

  // 1. concat + input projection
  k_concat<<<2048, 256, 0, stream>>>(ce, spk, xin);
  gemm_wmma<<<dim3(32, 8, 1), 256, 0, stream>>>(xin, 256, 0, in_w, 256, 0, in_b,
                                                x0, 256, 0, ROWS_, 256, 256,
                                                BIG, 0, 0);
  // 2. two LConv blocks
  const float* Vin = x0;
  float* Vout = V1;
  for (int i = 0; i < 2; ++i) {
    const int base = 7 + i * 7;
    const float* lg  = (const float*)d_in[base + 0];
    const float* lb  = (const float*)d_in[base + 1];
    const float* gw  = (const float*)d_in[base + 2];
    const float* gb  = (const float*)d_in[base + 3];
    const float* dw  = (const float*)d_in[base + 4];
    const float* ow  = (const float*)d_in[base + 5];
    const float* ob  = (const float*)d_in[base + 6];
    k_ln256<<<ROWS_, 256, 0, stream>>>(Vin, lnbuf, lg, lb);
    gemm_wmma<<<dim3(32, 16, 1), 256, 0, stream>>>(lnbuf, 256, 0, gw, 512, 0,
                                                   gb, hglu, 512, 0, ROWS_, 512,
                                                   256, BIG, 0, 0);
    k_glu<<<2048, 256, 0, stream>>>(hglu, tmask, gbuf);
    k_dwconv<<<2048, 256, 0, stream>>>(gbuf, dw, g2buf);
    k_copy<<<2048, 256, 0, stream>>>(Vin, Vout, (int)RC);   // residual
    gemm_wmma<<<dim3(32, 8, 1), 256, 0, stream>>>(g2buf, 256, 0, ow, 256, 0, ob,
                                                  Vout, 256, 0, ROWS_, 256, 256,
                                                  BIG, 0, 1);
    k_maskrows<<<2048, 256, 0, stream>>>(Vout, tmask);
    Vin = Vout; Vout = V2;
  }
  const float* Vfin = Vin;   // output of second LConv block (== V2)
  // 3. duration head + T = conv1d(V, p2)
  k_p1<<<ROWS_, 256, 0, stream>>>(Vfin, p1_w, p1_b, tmask, Dv);
  for (int kw = 0; kw < 3; ++kw)
    gemm_wmma<<<dim3(32, 8, 1), 256, 0, stream>>>(
        Vfin, 256, 0, p2_k + (size_t)kw * 256 * 256, 256, 0,
        (kw == 0) ? p2_b : nullptr, T, 256, 0, ROWS_, 256, 256, 256, kw - 1,
        kw > 0);
  // 4. per-batch scalars / scan / masks / scaling
  k_scalars<<<B_, 256, 0, stream>>>(Dv, sched, rhythm, Vfin, T, ek, sk, sV, sT,
                                    mel_i, out_a, out_ml, out_mm);
  // 5. Cv = LN(relu(conv1d(sV, cw)))
  for (int kw = 0; kw < 3; ++kw)
    gemm_wmma<<<dim3(32, 1, 1), 256, 0, stream>>>(
        sV, 256, 0, cw_k + (size_t)kw * 256 * 16, 16, 0,
        (kw == 0) ? cw_b : nullptr, Cvraw, 16, 0, ROWS_, 16, 256, 256, kw - 1,
        kw > 0);
  k_reluln16<<<8, 256, 0, stream>>>(Cvraw, Cv, cw_ln_g, cw_ln_b);
  // 6. the two heavy 2-D convs (implicit GEMM, WMMA), conv2 fused with lw
  k_conv1_wmma<<<dim3(16, 125, B_), 256, 0, stream>>>(sk, ek, Cv, tmask, mel_i,
                                                      mkb1_k, mkb1_b, h1);
  k_conv2_wmma<<<dim3(16, 125, B_), 256, 0, stream>>>(h1, mkb2_k, mkb2_b, lw_w,
                                                      lw_b, Wsc);
  // 7. softmax over S -> Wmat (output + ws copy for the matmul)
  k_softmax<<<dim3(MEL_, B_), 256, 0, stream>>>(Wsc, tmask, out_W, Wsc);
  // 8. up = Wmat @ sT  (batched WMMA GEMM), then LN + mel mask
  gemm_wmma<<<dim3(16, 8, B_), 256, 0, stream>>>(
      Wsc, 256, (long long)MEL_ * 256, sT, 256, 65536LL, nullptr, upraw, 256,
      (long long)MEL_ * 256, MEL_, 256, 256, BIG, 0, 0);
  k_upln<<<dim3(MEL_, B_), 256, 0, stream>>>(upraw, ln_g, ln_b, mel_i, out_up);
}
